// AttentionBlock_55783035240614
// MI455X (gfx1250) — matrix operand order repaired and checked by reading, not compiled, not run on hardware
//
#include <hip/hip_runtime.h>
#include <hip/hip_bf16.h>

// ---------------------------------------------------------------------------
// Sizes from the reference
#define NB   4      // batch
#define CCH  256    // channels
#define DQK  32     // q/k channels
#define NN   4096   // H*W
// ---------------------------------------------------------------------------

typedef __attribute__((ext_vector_type(16))) __bf16 v16bf;
typedef __attribute__((ext_vector_type(8)))  float  v8f;

union BfPack {
  unsigned int u[8];
  v16bf        v;
};

// round-half-up fp32 -> bf16 (1 VALU)
__device__ __forceinline__ unsigned int rnd(float f) {
  return __builtin_bit_cast(unsigned int, f) + 0x8000u;
}
// pack two fp32 into {bf16(hi),bf16(lo)} with a single v_perm_b32
__device__ __forceinline__ unsigned int pk2(float lo, float hi) {
  return __builtin_amdgcn_perm(rnd(hi), rnd(lo), 0x07060302u);
}
__device__ __forceinline__ unsigned short f2bf(float f) {
  return (unsigned short)(rnd(f) >> 16);
}
__device__ __forceinline__ v8f wmma_bf16(const BfPack& a, const BfPack& b, v8f c) {
  // D = A(16x32 bf16) x B(32x16 bf16) + C(16x16 f32)
  return __builtin_amdgcn_wmma_f32_16x16x32_bf16(false, a.v, false, b.v,
                                                 (short)0, c, false, false);
}

// ===========================================================================
// Kernel 1: QKV projection (1x1 convs), fp32 -> bf16.
//   Qb[b][n][d], Kb[b][n][d]  (row-major, d=32)  -> direct WMMA A/B operands
//   Vc[b][c][m]               (row-major, m=N)   -> direct WMMA B operand
// One block = one 16-wide n tile, 4 waves share the staged x^T tile in LDS.
// ===========================================================================
__global__ __launch_bounds__(128, 1)
void qkv_proj_kernel(const float* __restrict__ x,
                     const float* __restrict__ Wq, const float* __restrict__ bq,
                     const float* __restrict__ Wk, const float* __restrict__ bk,
                     const float* __restrict__ Wv, const float* __restrict__ bv,
                     unsigned short* __restrict__ qb,
                     unsigned short* __restrict__ kb,
                     unsigned short* __restrict__ vc) {
  __shared__ unsigned short XT[16 * CCH];        // x^T tile: [n_local][cin] bf16
  unsigned int* XT32 = (unsigned int*)XT;

  const int tid = threadIdx.x;
  const int b   = blockIdx.x / (NN / 16);
  const int n0  = (blockIdx.x % (NN / 16)) * 16;

  // ---- stage x^T (bf16) into LDS; each thread owns one channel *pair* so
  //      each (row, c-pair) becomes one v_perm_b32 + one ds_store_b32 ----
  const float* xb = x + (size_t)b * CCH * NN;
  {
    const int cp = tid;                          // 128 pairs, one per thread
    const int c  = cp * 2;
    const float* s0 = xb + (size_t)c * NN + n0;  // 64B aligned
    const float* s1 = s0 + NN;
    const float4 a0 = ((const float4*)s0)[0], a1 = ((const float4*)s0)[1];
    const float4 a2 = ((const float4*)s0)[2], a3 = ((const float4*)s0)[3];
    const float4 b0 = ((const float4*)s1)[0], b1 = ((const float4*)s1)[1];
    const float4 b2 = ((const float4*)s1)[2], b3 = ((const float4*)s1)[3];
    const float va[16] = {a0.x,a0.y,a0.z,a0.w, a1.x,a1.y,a1.z,a1.w,
                          a2.x,a2.y,a2.z,a2.w, a3.x,a3.y,a3.z,a3.w};
    const float vb[16] = {b0.x,b0.y,b0.z,b0.w, b1.x,b1.y,b1.z,b1.w,
                          b2.x,b2.y,b2.z,b2.w, b3.x,b3.y,b3.z,b3.w};
#pragma unroll
    for (int i = 0; i < 16; ++i)
      XT32[i * (CCH / 2) + cp] = pk2(va[i], vb[i]);
  }
  __syncthreads();

  const int wid  = tid >> 5;
  const int lane = tid & 31;
  const int r    = lane & 15;      // 0..15
  const int h    = lane >> 4;      // lane half
  const char* XTb = (const char*)XT;

  // 20 output tiles: j=0,1 -> Q, j=2,3 -> K, j=4..19 -> V
  for (int j = wid; j < 20; j += 4) {
    if (j < 4) {
      // --- Q/K: D[n][o] = sum_c x^T[n][c] * W^T[c][o] ---
      const float* W     = (j < 2) ? Wq : Wk;
      const float* bias  = (j < 2) ? bq : bk;
      unsigned short* dst = ((j < 2) ? qb : kb) + (size_t)b * NN * DQK;
      const int ot = (j & 1) * 16;

      v8f acc = {};
#pragma unroll
      for (int cc = 0; cc < CCH; cc += 32) {
        // A operand from LDS (rows = n). Lane half h selects contiguous K half.
        BfPack a, bb;
        const int off0 = r * (CCH * 2) + cc * 2 + h * 32;
        uint4 a0 = *(const uint4*)(XTb + off0);
        uint4 a1 = *(const uint4*)(XTb + off0 + 16);
        a.u[0]=a0.x; a.u[1]=a0.y; a.u[2]=a0.z; a.u[3]=a0.w;
        a.u[4]=a1.x; a.u[5]=a1.y; a.u[6]=a1.z; a.u[7]=a1.w;
        // B operand = W^T column o = ot+r: contiguous 16 fp32 of W row (ot+r)
        const float* wrow = W + (size_t)(ot + r) * CCH + cc + h * 16;
        float4 w0 = ((const float4*)wrow)[0];
        float4 w1 = ((const float4*)wrow)[1];
        float4 w2 = ((const float4*)wrow)[2];
        float4 w3 = ((const float4*)wrow)[3];
        bb.u[0]=pk2(w0.x,w0.y); bb.u[1]=pk2(w0.z,w0.w);
        bb.u[2]=pk2(w1.x,w1.y); bb.u[3]=pk2(w1.z,w1.w);
        bb.u[4]=pk2(w2.x,w2.y); bb.u[5]=pk2(w2.z,w2.w);
        bb.u[6]=pk2(w3.x,w3.y); bb.u[7]=pk2(w3.z,w3.w);
        acc = wmma_bf16(a, bb, acc);
      }
      const float bo = bias[ot + r];
#pragma unroll
      for (int v = 0; v < 8; ++v) {
        const int n = n0 + v + 8 * h;                 // D row (C/D layout)
        dst[(size_t)n * DQK + ot + r] = f2bf(acc[v] + bo);
      }
    } else {
      // --- V: D[c][n] = sum_cin Wv[c][cin] * x[cin][n] ---
      const int ot = (j - 4) * 16;
      const float* wrow = Wv + (size_t)(ot + r) * CCH;  // A row c = ot+r
      v8f acc = {};
#pragma unroll
      for (int cc = 0; cc < CCH; cc += 32) {
        BfPack a, bb;
        // A operand = Wv rows; K interleave per lane half
        const float* pa = wrow + cc + h * 8;
        float4 w0 = ((const float4*)pa)[0];
        float4 w1 = ((const float4*)pa)[1];
        const float* pc = wrow + cc + 16 + h * 8;
        float4 w2 = ((const float4*)pc)[0];
        float4 w3 = ((const float4*)pc)[1];
        a.u[0]=pk2(w0.x,w0.y); a.u[1]=pk2(w0.z,w0.w);
        a.u[2]=pk2(w1.x,w1.y); a.u[3]=pk2(w1.z,w1.w);
        a.u[4]=pk2(w2.x,w2.y); a.u[5]=pk2(w2.z,w2.w);
        a.u[6]=pk2(w3.x,w3.y); a.u[7]=pk2(w3.z,w3.w);
        // B operand = x tile: column n = n0+r -> LDS row r, K = cc+h*8..+7, cc+16+h*8..+7
        const int off = r * (CCH * 2) + (cc + h * 8) * 2;
        uint4 b0 = *(const uint4*)(XTb + off);
        uint4 b1 = *(const uint4*)(XTb + off + 32);
        bb.u[0]=b0.x; bb.u[1]=b0.y; bb.u[2]=b0.z; bb.u[3]=b0.w;
        bb.u[4]=b1.x; bb.u[5]=b1.y; bb.u[6]=b1.z; bb.u[7]=b1.w;
        acc = wmma_bf16(a, bb, acc);
      }
#pragma unroll
      for (int v = 0; v < 8; ++v) {
        const int c = ot + v + 8 * h;                 // D row (C/D layout)
        vc[((size_t)b * CCH + c) * NN + n0 + r] = f2bf(acc[v] + bv[c]);
      }
    }
  }
}

// ===========================================================================
// Kernel 2: flash attention, one wave per 16 query rows, 128-wide KV chunks.
//   Per chunk: 8x WMMA S = Q K^T, online softmax (shfl_xor reductions in
//   16-lane halves, amortized over 128 cols), P -> per-wave LDS -> 4 A
//   operands, 64x WMMA for O += P V.  Epilogue: out = gamma*O/rowsum + x.
// ===========================================================================
#define MCW 128                                  // KV chunk width
__global__ __launch_bounds__(128, 1)
void attn_kernel(const float* __restrict__ x, const float* __restrict__ gamma,
                 const unsigned short* __restrict__ qb,
                 const unsigned short* __restrict__ kb,
                 const unsigned short* __restrict__ vc,
                 float* __restrict__ out) {
  __shared__ unsigned short P[4 * 16 * MCW];     // 4 KB per wave, 16 KB total

  const int tid  = threadIdx.x;
  const int wid  = tid >> 5;
  const int lane = tid & 31;
  const int r    = lane & 15;
  const int h    = lane >> 4;
  const int gidx = blockIdx.x * 4 + wid;
  const int b    = gidx / (NN / 16);
  const int n0   = (gidx % (NN / 16)) * 16;

  unsigned short* Pw = P + wid * (16 * MCW);
  const char* Pb = (const char*)Pw;

  // Q A-operand (reused for every KV chunk)
  BfPack aq;
  {
    const char* qrow = (const char*)(qb + ((size_t)b * NN + n0 + r) * DQK);
    uint4 q0 = *(const uint4*)(qrow + h * 32);
    uint4 q1 = *(const uint4*)(qrow + h * 32 + 16);
    aq.u[0]=q0.x; aq.u[1]=q0.y; aq.u[2]=q0.z; aq.u[3]=q0.w;
    aq.u[4]=q1.x; aq.u[5]=q1.y; aq.u[6]=q1.z; aq.u[7]=q1.w;
  }

  v8f acc[16];
#pragma unroll
  for (int i = 0; i < 16; ++i) acc[i] = (v8f){0,0,0,0,0,0,0,0};
  float mi[8], li[8];
#pragma unroll
  for (int v = 0; v < 8; ++v) { mi[v] = -3.0e38f; li[v] = 0.0f; }

  const unsigned short* kbb = kb + (size_t)b * NN * DQK;
  const unsigned short* vcb = vc + (size_t)b * CCH * NN;

  for (int mc = 0; mc < NN; mc += MCW) {
    // ---- scores: 8 tiles of S = Q K^T ----
    v8f s[8];
#pragma unroll
    for (int t = 0; t < 8; ++t) {
      BfPack bk;
      const char* krow =
          (const char*)(kbb + (size_t)(mc + 16 * t + r) * DQK) + h * 32;
      uint4 k0 = *(const uint4*)(krow);
      uint4 k1 = *(const uint4*)(krow + 16);
      bk.u[0]=k0.x; bk.u[1]=k0.y; bk.u[2]=k0.z; bk.u[3]=k0.w;
      bk.u[4]=k1.x; bk.u[5]=k1.y; bk.u[6]=k1.z; bk.u[7]=k1.w;
      v8f z = {};
      s[t] = wmma_bf16(aq, bk, z);
    }

    // ---- online softmax over this 128-wide chunk ----
    v8f alphav;
#pragma unroll
    for (int v = 0; v < 8; ++v) {
      float cm = fmaxf(fmaxf(fmaxf(s[0][v], s[1][v]), fmaxf(s[2][v], s[3][v])),
                       fmaxf(fmaxf(s[4][v], s[5][v]), fmaxf(s[6][v], s[7][v])));
      cm = fmaxf(cm, __shfl_xor(cm, 1, 32));
      cm = fmaxf(cm, __shfl_xor(cm, 2, 32));
      cm = fmaxf(cm, __shfl_xor(cm, 4, 32));
      cm = fmaxf(cm, __shfl_xor(cm, 8, 32));
      const float mn = fmaxf(mi[v], cm);
      alphav[v] = __expf(mi[v] - mn);
      float rs = 0.0f;
#pragma unroll
      for (int t = 0; t < 8; ++t) {
        const float p = __expf(s[t][v] - mn);
        s[t][v] = p;
        rs += p;
      }
      rs += __shfl_xor(rs, 1, 32);
      rs += __shfl_xor(rs, 2, 32);
      rs += __shfl_xor(rs, 4, 32);
      rs += __shfl_xor(rs, 8, 32);
      li[v] = li[v] * alphav[v] + rs;
      mi[v] = mn;
    }
    // rescale running output accumulators (vector form -> packed f32 muls)
#pragma unroll
    for (int ct = 0; ct < 16; ++ct) acc[ct] *= alphav;

    // ---- P (bf16) through per-wave LDS to re-layout into A operands ----
#pragma unroll
    for (int t = 0; t < 8; ++t)
#pragma unroll
      for (int v = 0; v < 8; ++v)
        Pw[(v + 8 * h) * MCW + 16 * t + r] = f2bf(s[t][v]);

    // same-wave LDS store->load ordering
    asm volatile("s_wait_dscnt 0" ::: "memory");

    BfPack ap[4];
    {
      const char* prow = Pb + r * (MCW * 2);
#pragma unroll
      for (int k = 0; k < 4; ++k) {
        uint4 p0 = *(const uint4*)(prow + k * 64 + h * 16);
        uint4 p1 = *(const uint4*)(prow + k * 64 + 32 + h * 16);
        ap[k].u[0]=p0.x; ap[k].u[1]=p0.y; ap[k].u[2]=p0.z; ap[k].u[3]=p0.w;
        ap[k].u[4]=p1.x; ap[k].u[5]=p1.y; ap[k].u[6]=p1.z; ap[k].u[7]=p1.w;
      }
    }

    // ---- O += P * V over all 256 channels (16 c tiles x 4 K-quarters) ----
#pragma unroll
    for (int ct = 0; ct < 16; ++ct) {
      const char* vrow =
          (const char*)(vcb + (size_t)(ct * 16 + r) * NN + mc);
#pragma unroll
      for (int k = 0; k < 4; ++k) {
        BfPack bv;
        uint4 u0 = *(const uint4*)(vrow + k * 64 + h * 16);
        uint4 u1 = *(const uint4*)(vrow + k * 64 + h * 16 + 32);
        bv.u[0]=u0.x; bv.u[1]=u0.y; bv.u[2]=u0.z; bv.u[3]=u0.w;
        bv.u[4]=u1.x; bv.u[5]=u1.y; bv.u[6]=u1.z; bv.u[7]=u1.w;
        acc[ct] = wmma_bf16(ap[k], bv, acc[ct]);
      }
    }
  }

  // ---- epilogue: out = gamma * O / l + x ----
  const float g = gamma[0];
  const float* xb = x + (size_t)b * CCH * NN;
  float* ob = out + (size_t)b * CCH * NN;
  v8f sc;
#pragma unroll
  for (int v = 0; v < 8; ++v) sc[v] = g / li[v];
#pragma unroll
  for (int ct = 0; ct < 16; ++ct) {
    const int c = ct * 16 + r;
    const v8f o = acc[ct] * sc;
#pragma unroll
    for (int v = 0; v < 8; ++v) {
      const int n = n0 + v + 8 * h;
      const size_t idx = (size_t)c * NN + n;
      ob[idx] = o[v] + xb[idx];
    }
  }
}

// ===========================================================================
extern "C" void kernel_launch(void* const* d_in, const int* in_sizes, int n_in,
                              void* d_out, int out_size, void* d_ws, size_t ws_size,
                              hipStream_t stream) {
  const float* x  = (const float*)d_in[0];
  const float* Wq = (const float*)d_in[1];
  const float* bq = (const float*)d_in[2];
  const float* Wk = (const float*)d_in[3];
  const float* bk = (const float*)d_in[4];
  const float* Wv = (const float*)d_in[5];
  const float* bv = (const float*)d_in[6];
  const float* gm = (const float*)d_in[7];
  float* out = (float*)d_out;

  // bf16 workspace: Q (1MB) | K (1MB) | V^ [b][c][m] (8MB)  = 10MB total
  unsigned short* qbp = (unsigned short*)d_ws;
  unsigned short* kbp = qbp + (size_t)NB * NN * DQK;
  unsigned short* vcp = kbp + (size_t)NB * NN * DQK;

  qkv_proj_kernel<<<dim3(NB * (NN / 16)), dim3(128), 0, stream>>>(
      x, Wq, bq, Wk, bk, Wv, bv, qbp, kbp, vcp);
  attn_kernel<<<dim3(NB * (NN / 16) / 4), dim3(128), 0, stream>>>(
      x, gm, qbp, kbp, vcp, out);
}
